// MambaMixerHuggingFace_22393959481415
// MI455X (gfx1250) — matrix operand order repaired and checked by reading, not compiled, not run on hardware
//
#include <hip/hip_runtime.h>
#include <hip/hip_bf16.h>

// ---------------- problem constants ----------------
constexpr int Bb  = 2;
constexpr int Ls  = 4096;
constexpr int Hh  = 1024;
constexpr int EDc = 2048;
constexpr int Nn  = 16;
constexpr int Rr  = 64;
constexpr int BM  = Bb * Ls;          // 8192 rows for all GEMMs
constexpr int CHUNK  = 128;           // scan chunk length
constexpr int NCHUNK = Ls / CHUNK;    // 32

typedef __hip_bfloat16 bf16;

// ---------------- WMMA types ----------------
typedef __bf16 v16bf __attribute__((ext_vector_type(16)));
typedef float  v8f   __attribute__((ext_vector_type(8)));

struct alignas(16) U4 { unsigned a, b, c, d; };
union FragBF { v16bf v; U4 q[2]; };

__device__ inline v8f zero_v8f() {
  v8f z;
#pragma unroll
  for (int i = 0; i < 8; ++i) z[i] = 0.f;
  return z;
}

// A fragment: 16x32 bf16, row m = lane&15, K-half kh = 8*(lane>=16)
// elements 0..7  -> K = kk+kh .. kk+kh+7
// elements 8..15 -> K = kk+kh+16 .. kk+kh+23
__device__ inline FragBF load_a_frag(const bf16* __restrict__ A, size_t row, int Kdim,
                                     int kk, int lane) {
  FragBF f;
  const int m  = lane & 15;
  const int kh = (lane >> 4) * 8;
  const bf16* p = A + (row + m) * (size_t)Kdim + kk + kh;
  f.q[0] = *reinterpret_cast<const U4*>(p);
  f.q[1] = *reinterpret_cast<const U4*>(p + 16);
  return f;
}

// B fragment: 32x16 bf16 (K x N). W stored [N,K] row-major; per lane row n,
// elements 0..7 -> K = kk+kh..+7, elements 8..15 -> K = kk+kh+16..+23, kh = 8*(lane>=16).
__device__ inline FragBF load_b_frag(const bf16* __restrict__ W, size_t ncol0, int Kdim,
                                     int kk, int lane) {
  FragBF f;
  const int n  = lane & 15;
  const int kh = (lane >> 4) * 8;
  const bf16* p = W + (ncol0 + n) * (size_t)Kdim + kk + kh;
  f.q[0] = *reinterpret_cast<const U4*>(p);
  f.q[1] = *reinterpret_cast<const U4*>(p + 16);
  return f;
}

// ---------------- generic cast kernel ----------------
__global__ void cast_f32_bf16(const float* __restrict__ in, bf16* __restrict__ out, int n) {
  int i = blockIdx.x * blockDim.x + threadIdx.x;
  if (i < n) out[i] = __float2bfloat16(in[i]);
}

// ---------------- big GEMM: C[M,N] = A[M,K] * W[N,K]^T (bf16 in, f32 out) ----
// block = 128 threads = 4 waves; each wave owns a 64x64 tile.
// grid.x = N/256, grid.y = M/64. Kdim must be a multiple of 64.
// Double-buffered K loop: loads for the next 32-K slab are in flight while the
// current slab's 16 WMMAs issue, so waits are partial instead of full stalls.
template <bool SPLIT>
__global__ __launch_bounds__(128) void gemm_bf16(
    const bf16* __restrict__ A, const bf16* __restrict__ W,
    float* __restrict__ out0, float* __restrict__ out1,
    int Kdim, int ldc, int splitCol) {
  const int lane = threadIdx.x & 31;
  const int wave = threadIdx.x >> 5;
  const size_t n0 = (size_t)blockIdx.x * 256 + wave * 64;
  const size_t m0 = (size_t)blockIdx.y * 64;

  v8f acc[4][4];
#pragma unroll
  for (int i = 0; i < 4; ++i)
#pragma unroll
    for (int j = 0; j < 4; ++j) acc[i][j] = zero_v8f();

  FragBF a0[4], b0[4], a1[4], b1[4];
#pragma unroll
  for (int i = 0; i < 4; ++i) {
    a0[i] = load_a_frag(A, m0 + i * 16, Kdim, 0, lane);
    b0[i] = load_b_frag(W, n0 + i * 16, Kdim, 0, lane);
  }

  for (int kk = 0; kk < Kdim; kk += 64) {
    // prefetch slab kk+32 while computing slab kk
#pragma unroll
    for (int i = 0; i < 4; ++i) {
      a1[i] = load_a_frag(A, m0 + i * 16, Kdim, kk + 32, lane);
      b1[i] = load_b_frag(W, n0 + i * 16, Kdim, kk + 32, lane);
    }
#pragma unroll
    for (int i = 0; i < 4; ++i)
#pragma unroll
      for (int j = 0; j < 4; ++j)
        acc[i][j] = __builtin_amdgcn_wmma_f32_16x16x32_bf16(
            false, a0[i].v, false, b0[j].v, (short)0, acc[i][j], false, false);

    // prefetch slab kk+64 while computing slab kk+32
    if (kk + 64 < Kdim) {
#pragma unroll
      for (int i = 0; i < 4; ++i) {
        a0[i] = load_a_frag(A, m0 + i * 16, Kdim, kk + 64, lane);
        b0[i] = load_b_frag(W, n0 + i * 16, Kdim, kk + 64, lane);
      }
    }
#pragma unroll
    for (int i = 0; i < 4; ++i)
#pragma unroll
      for (int j = 0; j < 4; ++j)
        acc[i][j] = __builtin_amdgcn_wmma_f32_16x16x32_bf16(
            false, a1[i].v, false, b1[j].v, (short)0, acc[i][j], false, false);
  }

  const int rowhi = (lane >> 4) * 8;
  const int ncol  = lane & 15;
#pragma unroll
  for (int i = 0; i < 4; ++i)
#pragma unroll
    for (int j = 0; j < 4; ++j)
#pragma unroll
      for (int r = 0; r < 8; ++r) {
        const size_t row = m0 + i * 16 + r + rowhi;
        const size_t col = n0 + j * 16 + ncol;
        const float v = acc[i][j][r];
        if (SPLIT) {
          if ((int)col < splitCol) out0[row * ldc + col] = v;
          else                     out1[row * ldc + (col - splitCol)] = v;
        } else {
          out0[row * ldc + col] = v;
        }
      }
}

// ---------------- depthwise causal conv (K=4) + SiLU ----------------
// grid(ED/256, L/128, B), block 256 over e.
__global__ __launch_bounds__(256) void conv_silu(
    const float* __restrict__ raw, const float* __restrict__ cw,
    const float* __restrict__ cb, float* __restrict__ hs,
    bf16* __restrict__ hsbf) {
  const int e  = blockIdx.x * 256 + threadIdx.x;
  const int b  = blockIdx.z;
  const int l0 = blockIdx.y * 128;
  const float w0 = cw[e * 4 + 0], w1 = cw[e * 4 + 1], w2 = cw[e * 4 + 2], w3 = cw[e * 4 + 3];
  const float bias = cb[e];
  const size_t base = (size_t)b * Ls * EDc + e;
  float x1 = (l0 >= 1) ? raw[base + (size_t)(l0 - 1) * EDc] : 0.f;
  float x2 = (l0 >= 2) ? raw[base + (size_t)(l0 - 2) * EDc] : 0.f;
  float x3 = (l0 >= 3) ? raw[base + (size_t)(l0 - 3) * EDc] : 0.f;
  for (int l = l0; l < l0 + 128; ++l) {
    const size_t o = base + (size_t)l * EDc;
    const float x0 = raw[o];
    const float cv = bias + w0 * x3 + w1 * x2 + w2 * x1 + w3 * x0;
    const float s  = cv / (1.f + __expf(-cv));
    hs[o]   = s;
    hsbf[o] = __float2bfloat16(s);
    x3 = x2; x2 = x1; x1 = x0;
  }
}

// ---------------- x_proj GEMM: ssm[M,96] = hs_bf[M,2048] * Wxp[96,2048]^T ----
// block = 256 threads = 8 waves; each wave owns K-chunk of 256; LDS reduce.
// grid.x = M/16. B fragments are batch-loaded per K step and the 8-step loop
// is fully unrolled so loads can hoist over the previous step's WMMAs.
__global__ __launch_bounds__(256) void gemm_xproj(
    const bf16* __restrict__ hsbf, const bf16* __restrict__ wxp,
    float* __restrict__ ssm, bf16* __restrict__ dtrbf) {
  __shared__ float lds[16 * 96];
  const int tid  = threadIdx.x;
  const int lane = tid & 31;
  const int wave = tid >> 5;
  const size_t m0 = (size_t)blockIdx.x * 16;

  for (int i = tid; i < 16 * 96; i += 256) lds[i] = 0.f;
  __syncthreads();

  v8f acc[6];
#pragma unroll
  for (int t = 0; t < 6; ++t) acc[t] = zero_v8f();

  const int kw0 = wave * 256;
#pragma unroll
  for (int s = 0; s < 8; ++s) {
    const int kk = kw0 + s * 32;
    FragBF a = load_a_frag(hsbf, m0, EDc, kk, lane);
    FragBF b[6];
#pragma unroll
    for (int t = 0; t < 6; ++t) b[t] = load_b_frag(wxp, (size_t)t * 16, EDc, kk, lane);
#pragma unroll
    for (int t = 0; t < 6; ++t)
      acc[t] = __builtin_amdgcn_wmma_f32_16x16x32_bf16(
          false, a.v, false, b[t].v, (short)0, acc[t], false, false);
  }

  const int rowhi = (lane >> 4) * 8;
  const int ncol  = lane & 15;
#pragma unroll
  for (int t = 0; t < 6; ++t)
#pragma unroll
    for (int r = 0; r < 8; ++r)
      atomicAdd(&lds[(r + rowhi) * 96 + t * 16 + ncol], acc[t][r]);
  __syncthreads();

  for (int i = tid; i < 16 * 96; i += 256) {
    const int row = i / 96, col = i % 96;
    const float v = lds[i];
    ssm[(m0 + row) * 96 + col] = v;
    if (col < Rr) dtrbf[(m0 + row) * Rr + col] = __float2bfloat16(v);
  }
}

// ---------------- dt_proj GEMM + bias + softplus -> delta[M,2048] ----------
// block = 128 = 4 waves; wave tile 16x64. grid(M/16, N/256).
__global__ __launch_bounds__(128) void gemm_dtproj(
    const bf16* __restrict__ dtrbf, const bf16* __restrict__ wdt,
    const float* __restrict__ dt_b, float* __restrict__ delta) {
  const int lane = threadIdx.x & 31;
  const int wave = threadIdx.x >> 5;
  const size_t m0 = (size_t)blockIdx.x * 16;
  const size_t n0 = (size_t)blockIdx.y * 256 + wave * 64;

  v8f acc[4];
#pragma unroll
  for (int j = 0; j < 4; ++j) acc[j] = zero_v8f();

#pragma unroll
  for (int kk = 0; kk < Rr; kk += 32) {
    FragBF a = load_a_frag(dtrbf, m0, Rr, kk, lane);
    FragBF b[4];
#pragma unroll
    for (int j = 0; j < 4; ++j) b[j] = load_b_frag(wdt, n0 + j * 16, Rr, kk, lane);
#pragma unroll
    for (int j = 0; j < 4; ++j)
      acc[j] = __builtin_amdgcn_wmma_f32_16x16x32_bf16(
          false, a.v, false, b[j].v, (short)0, acc[j], false, false);
  }

  const int rowhi = (lane >> 4) * 8;
  const int ncol  = lane & 15;
#pragma unroll
  for (int j = 0; j < 4; ++j)
#pragma unroll
    for (int r = 0; r < 8; ++r) {
      const size_t col = n0 + j * 16 + ncol;
      const size_t row = m0 + r + rowhi;
      const float v  = acc[j][r] + dt_b[col];
      const float sp = (v > 20.f) ? v : log1pf(__expf(v));
      delta[row * EDc + col] = sp;
    }
}

// ---------------- selective scan, pass 1: per-chunk local state + decay sum -
// grid(ED/256, NCHUNK, B), block 256 over e.
__global__ __launch_bounds__(256) void scan_pass1(
    const float* __restrict__ delta, const float* __restrict__ hs,
    const float* __restrict__ ssm, const float* __restrict__ A_log,
    float* __restrict__ hout, float* __restrict__ Ssum) {
  const int e = blockIdx.x * 256 + threadIdx.x;
  const int c = blockIdx.y;
  const int b = blockIdx.z;
  float a[16], h[16];
#pragma unroll
  for (int n = 0; n < 16; ++n) { a[n] = -__expf(A_log[e * 16 + n]); h[n] = 0.f; }
  float S = 0.f;
  const int t0 = c * CHUNK;
  for (int t = t0; t < t0 + CHUNK; ++t) {
    const size_t idx = (size_t)b * Ls + t;
    const float d = delta[idx * EDc + e];
    const float u = hs[idx * EDc + e];
    const float* Bp = ssm + idx * 96 + 64;
    const float du = d * u;
    S += d;
#pragma unroll
    for (int n = 0; n < 16; ++n)
      h[n] = __expf(d * a[n]) * h[n] + du * Bp[n];
  }
  const size_t ch = (size_t)b * EDc + e;
#pragma unroll
  for (int n = 0; n < 16; ++n) hout[(ch * NCHUNK + c) * 16 + n] = h[n];
  Ssum[ch * NCHUNK + c] = S;
}

// ---------------- pass 2: carry combine (sequential over 32 chunks) --------
__global__ __launch_bounds__(256) void scan_pass2(
    const float* __restrict__ hout, const float* __restrict__ Ssum,
    const float* __restrict__ A_log, float* __restrict__ hin) {
  const int gid = blockIdx.x * 256 + threadIdx.x;  // 0 .. B*ED*16-1
  const int n  = gid & 15;
  const int ch = gid >> 4;
  const int e  = ch & (EDc - 1);
  const float a = -__expf(A_log[e * 16 + n]);
  float h = 0.f;
  for (int c = 0; c < NCHUNK; ++c) {
    const size_t o = ((size_t)ch * NCHUNK + c) * 16 + n;
    hin[o] = h;
    h = __expf(a * Ssum[(size_t)ch * NCHUNK + c]) * h + hout[o];
  }
}

// ---------------- pass 3: replay chunk with carried state, fuse epilogue ---
__global__ __launch_bounds__(256) void scan_pass3(
    const float* __restrict__ delta, const float* __restrict__ hs,
    const float* __restrict__ ssm, const float* __restrict__ A_log,
    const float* __restrict__ hin, const float* __restrict__ Dvec,
    const float* __restrict__ gate, bf16* __restrict__ ybf) {
  const int e = blockIdx.x * 256 + threadIdx.x;
  const int c = blockIdx.y;
  const int b = blockIdx.z;
  const size_t ch = (size_t)b * EDc + e;
  float a[16], h[16];
#pragma unroll
  for (int n = 0; n < 16; ++n) {
    a[n] = -__expf(A_log[e * 16 + n]);
    h[n] = hin[(ch * NCHUNK + c) * 16 + n];
  }
  const float Dv = Dvec[e];
  const int t0 = c * CHUNK;
  for (int t = t0; t < t0 + CHUNK; ++t) {
    const size_t idx = (size_t)b * Ls + t;
    const float d = delta[idx * EDc + e];
    const float u = hs[idx * EDc + e];
    const float* Bp = ssm + idx * 96 + 64;
    const float* Cp = Bp + 16;
    const float du = d * u;
    float y0 = 0.f, y1 = 0.f, y2 = 0.f, y3 = 0.f;
#pragma unroll
    for (int n = 0; n < 4; ++n) {
      h[n]      = __expf(d * a[n])      * h[n]      + du * Bp[n];      y0 = fmaf(h[n],      Cp[n],      y0);
      h[n + 4]  = __expf(d * a[n + 4])  * h[n + 4]  + du * Bp[n + 4];  y1 = fmaf(h[n + 4],  Cp[n + 4],  y1);
      h[n + 8]  = __expf(d * a[n + 8])  * h[n + 8]  + du * Bp[n + 8];  y2 = fmaf(h[n + 8],  Cp[n + 8],  y2);
      h[n + 12] = __expf(d * a[n + 12]) * h[n + 12] + du * Bp[n + 12]; y3 = fmaf(h[n + 12], Cp[n + 12], y3);
    }
    float y = (y0 + y1) + (y2 + y3) + Dv * u;
    const float g = gate[idx * EDc + e];
    y *= g / (1.f + __expf(-g));
    ybf[idx * EDc + e] = __float2bfloat16(y);
  }
}

// ---------------- workspace layout (bytes) ----------------
constexpr size_t OFF_XBF   = 0;                                   // 8192*1024 bf16
constexpr size_t OFF_WIN   = OFF_XBF   + (size_t)BM * Hh * 2;     // 4096*1024 bf16
constexpr size_t OFF_WXP   = OFF_WIN   + (size_t)2 * EDc * Hh * 2;
constexpr size_t OFF_WDT   = OFF_WXP   + 524288;                  // 96*2048 bf16 (393216, padded)
constexpr size_t OFF_WOUT  = OFF_WDT   + (size_t)EDc * Rr * 2;
constexpr size_t OFF_HSRAW = OFF_WOUT  + (size_t)Hh * EDc * 2;    // also aliased as DELTA
constexpr size_t OFF_GATE  = OFF_HSRAW + (size_t)BM * EDc * 4;
constexpr size_t OFF_HS    = OFF_GATE  + (size_t)BM * EDc * 4;
constexpr size_t OFF_HSBF  = OFF_HS    + (size_t)BM * EDc * 4;
constexpr size_t OFF_SSM   = OFF_HSBF  + (size_t)BM * EDc * 2;
constexpr size_t OFF_DTRBF = OFF_SSM   + (size_t)BM * 96 * 4;
constexpr size_t OFF_HOUT  = OFF_DTRBF + (size_t)BM * Rr * 2;
constexpr size_t OFF_SSUM  = OFF_HOUT  + (size_t)Bb * EDc * NCHUNK * Nn * 4;
constexpr size_t OFF_HIN   = OFF_SSUM  + (size_t)Bb * EDc * NCHUNK * 4;
constexpr size_t OFF_YBF   = OFF_HIN   + (size_t)Bb * EDc * NCHUNK * Nn * 4;
constexpr size_t OFF_DELTA = OFF_HSRAW;  // reuse: pre-conv buffer dead after conv

extern "C" void kernel_launch(void* const* d_in, const int* in_sizes, int n_in,
                              void* d_out, int out_size, void* d_ws, size_t ws_size,
                              hipStream_t stream) {
  (void)in_sizes; (void)n_in; (void)out_size; (void)ws_size;
  const float* x      = (const float*)d_in[0];
  const float* w_in   = (const float*)d_in[1];
  const float* conv_w = (const float*)d_in[2];
  const float* conv_b = (const float*)d_in[3];
  const float* w_xp   = (const float*)d_in[4];
  const float* w_dt   = (const float*)d_in[5];
  const float* dt_b   = (const float*)d_in[6];
  const float* A_log  = (const float*)d_in[7];
  const float* Dvec   = (const float*)d_in[8];
  const float* w_out  = (const float*)d_in[9];
  float* out = (float*)d_out;

  char* ws = (char*)d_ws;
  bf16*  xbf    = (bf16*) (ws + OFF_XBF);
  bf16*  winbf  = (bf16*) (ws + OFF_WIN);
  bf16*  wxpbf  = (bf16*) (ws + OFF_WXP);
  bf16*  wdtbf  = (bf16*) (ws + OFF_WDT);
  bf16*  woutbf = (bf16*) (ws + OFF_WOUT);
  float* hsraw  = (float*)(ws + OFF_HSRAW);
  float* gateb  = (float*)(ws + OFF_GATE);
  float* hsf    = (float*)(ws + OFF_HS);
  bf16*  hsbf   = (bf16*) (ws + OFF_HSBF);
  float* ssm    = (float*)(ws + OFF_SSM);
  bf16*  dtrbf  = (bf16*) (ws + OFF_DTRBF);
  float* hout   = (float*)(ws + OFF_HOUT);
  float* Ssum   = (float*)(ws + OFF_SSUM);
  float* hinb   = (float*)(ws + OFF_HIN);
  bf16*  ybf    = (bf16*) (ws + OFF_YBF);
  float* deltab = (float*)(ws + OFF_DELTA);

  // 1. cast activations + weights to bf16
  auto nb = [](int n) { return (n + 255) / 256; };
  cast_f32_bf16<<<nb(BM * Hh),       256, 0, stream>>>(x,     xbf,    BM * Hh);
  cast_f32_bf16<<<nb(2 * EDc * Hh),  256, 0, stream>>>(w_in,  winbf,  2 * EDc * Hh);
  cast_f32_bf16<<<nb(96 * EDc),      256, 0, stream>>>(w_xp,  wxpbf,  96 * EDc);
  cast_f32_bf16<<<nb(EDc * Rr),      256, 0, stream>>>(w_dt,  wdtbf,  EDc * Rr);
  cast_f32_bf16<<<nb(Hh * EDc),      256, 0, stream>>>(w_out, woutbf, Hh * EDc);

  // 2. in_proj GEMM (M=8192, N=4096, K=1024), split into hs_raw | gate
  gemm_bf16<true><<<dim3((2 * EDc) / 256, BM / 64), 128, 0, stream>>>(
      xbf, winbf, hsraw, gateb, Hh, EDc, EDc);

  // 3. causal depthwise conv + SiLU (fp32 + bf16 mirror)
  conv_silu<<<dim3(EDc / 256, Ls / 128, Bb), 256, 0, stream>>>(
      hsraw, conv_w, conv_b, hsf, hsbf);

  // 4. x_proj GEMM (M=8192, N=96, K=2048) with LDS K-split reduction
  gemm_xproj<<<BM / 16, 256, 0, stream>>>(hsbf, wxpbf, ssm, dtrbf);

  // 5. dt_proj GEMM + bias + softplus -> delta (aliases hs_raw)
  gemm_dtproj<<<dim3(BM / 16, EDc / 256), 128, 0, stream>>>(dtrbf, wdtbf, dt_b, deltab);

  // 6-8. chunked selective scan
  scan_pass1<<<dim3(EDc / 256, NCHUNK, Bb), 256, 0, stream>>>(
      deltab, hsf, ssm, A_log, hout, Ssum);
  scan_pass2<<<(Bb * EDc * Nn) / 256, 256, 0, stream>>>(hout, Ssum, A_log, hinb);
  scan_pass3<<<dim3(EDc / 256, NCHUNK, Bb), 256, 0, stream>>>(
      deltab, hsf, ssm, A_log, hinb, Dvec, gateb, ybf);

  // 9. out_proj GEMM (M=8192, N=1024, K=2048) -> fp32 output
  gemm_bf16<false><<<dim3(Hh / 256, BM / 64), 128, 0, stream>>>(
      ybf, woutbf, out, out, EDc, Hh, 1 << 30);
}
